// SparseTransformerAttention_32839319945664
// MI455X (gfx1250) — matrix operand order repaired and checked by reading, not compiled, not run on hardware
//
#include <hip/hip_runtime.h>

typedef __bf16 bf16;
typedef __attribute__((ext_vector_type(16))) __bf16 v16bf;
typedef __attribute__((ext_vector_type(8)))  __bf16 v8bf;
typedef __attribute__((ext_vector_type(8)))  float  v8f;

static constexpr int kBatch  = 2;
static constexpr int kSeq    = 2048;
static constexpr int kHid    = 1024;
static constexpr int kNH     = 16;
static constexpr int kHD     = 64;
static constexpr int kStride = 45;          // int(sqrt(2048))
static constexpr int kRows   = kBatch * kSeq;        // 4096
static constexpr size_t kXN  = (size_t)kRows * kHid; // 4,194,304
static constexpr size_t kWN  = (size_t)kHid * kHid;  // 1,048,576

// ---------------------------------------------------------------------------
// fp32 -> bf16 conversion (grid-stride)
// ---------------------------------------------------------------------------
__global__ void cvt_f32_bf16(const float* __restrict__ src, bf16* __restrict__ dst,
                             int n) {
  int i = blockIdx.x * blockDim.x + threadIdx.x;
  int st = gridDim.x * blockDim.x;
  for (; i < n; i += st) dst[i] = (bf16)src[i];
}

// ---------------------------------------------------------------------------
// WMMA operand loaders (wave32 CDNA5 layouts, bf16 16x16x32)
// A operand: lane holds row `row`; VGPR0-3 = K[sel..sel+7], VGPR4-7 = K[16+sel..]
//            with sel = (lane>=16)?8:0
// B operand: lane holds column (row of source since B = src^T); same K order
//            as A: K[sel..sel+7], K[16+sel..] with sel = (lane>=16)?8:0
// ---------------------------------------------------------------------------
__device__ __forceinline__ v16bf load_mat_a(const bf16* base, int stride,
                                            int row, int kofs, int lane) {
  int sel = ((lane >> 4) & 1) << 3;  // 0 or 8
  const bf16* p = base + (size_t)row * stride + kofs + sel;
  v8bf c0 = *(const v8bf*)(p);
  v8bf c1 = *(const v8bf*)(p + 16);
  v16bf a;
#pragma unroll
  for (int t = 0; t < 8; ++t) { a[t] = c0[t]; a[8 + t] = c1[t]; }
  return a;
}

__device__ __forceinline__ v16bf load_mat_b(const bf16* base, int stride,
                                            int row, int kofs, int lane) {
  int sel = ((lane >> 4) & 1) << 3;  // 0 or 8
  const bf16* p = base + (size_t)row * stride + kofs + sel;
  v8bf c0 = *(const v8bf*)(p);
  v8bf c1 = *(const v8bf*)(p + 16);
  v16bf b;
#pragma unroll
  for (int t = 0; t < 8; ++t) { b[t] = c0[t]; b[8 + t] = c1[t]; }
  return b;
}

#define WMMA_BF16(A, B, C) \
  __builtin_amdgcn_wmma_f32_16x16x32_bf16(false, (A), false, (B), (short)0, (C), false, false)

// async DMA of 16 bytes/lane from global to LDS (ASYNCcnt path, gfx1250)
__device__ __forceinline__ void async_copy_b128(unsigned lds_off, const bf16* gptr) {
  asm volatile("global_load_async_to_lds_b128 %0, %1, off"
               :: "v"(lds_off), "v"(gptr) : "memory");
}
__device__ __forceinline__ void wait_async0() {
  asm volatile("s_wait_asynccnt 0" ::: "memory");
}

// ---------------------------------------------------------------------------
// Projection GEMM, 2x4 register-blocked: each wave computes a 32x64 block of
// out = xb @ W^T. All 8 waves of a WG share the same 32-row A panel, so it is
// staged ONCE per WG into LDS with global_load_async_to_lds_b128 (8x less A
// traffic on the vector path), then consumed via ds_load. B stays on direct
// global b128 loads (unique per wave, shared across WGs through L2).
// mode 0: store bf16 as [b, h, s, d]   (q / k)
// mode 1: store bf16 as [b, h, d, s]   (v transposed for PV WMMA)
// ---------------------------------------------------------------------------
__global__ void __launch_bounds__(256) proj_gemm(const bf16* __restrict__ xb,
                                                 const bf16* __restrict__ w,
                                                 bf16* __restrict__ outb,
                                                 int mode, float scale) {
  __shared__ bf16 apanel[32 * 64];  // 4 KB K-panel, shared by the whole WG

  int tid  = threadIdx.x;
  int lane = tid & 31;
  int wave = tid >> 5;
  int gw   = blockIdx.x * 8 + wave;      // 0..2047
  int m0   = (gw >> 4) * 32;             // constant within a WG
  int n0   = (gw & 15) * 64;
  int lo = lane & 15, hi = lane >> 4;

  // staging assignment: 256 threads x 16B = 4 KB panel (row = tid/8, 16B-chunk)
  int srow = tid >> 3;
  int scol = (tid & 7) * 8;
  const bf16* sgp = xb + (size_t)(m0 + srow) * kHid + scol;
  unsigned slds = (unsigned)(uintptr_t)(apanel + srow * 64 + scol);

  v8f acc[2][4] = {};
  for (int ks = 0; ks < kHid; ks += 64) {
    async_copy_b128(slds, sgp + ks);
    wait_async0();
    __syncthreads();
#pragma unroll
    for (int kk = 0; kk < 64; kk += 32) {
      v16bf a0 = load_mat_a(apanel, 64, lo,      kk, lane);
      v16bf a1 = load_mat_a(apanel, 64, 16 + lo, kk, lane);
      v16bf b0 = load_mat_b(w, kHid, n0 + lo,      ks + kk, lane);
      v16bf b1 = load_mat_b(w, kHid, n0 + 16 + lo, ks + kk, lane);
      v16bf b2 = load_mat_b(w, kHid, n0 + 32 + lo, ks + kk, lane);
      v16bf b3 = load_mat_b(w, kHid, n0 + 48 + lo, ks + kk, lane);
      acc[0][0] = WMMA_BF16(a0, b0, acc[0][0]);
      acc[0][1] = WMMA_BF16(a0, b1, acc[0][1]);
      acc[0][2] = WMMA_BF16(a0, b2, acc[0][2]);
      acc[0][3] = WMMA_BF16(a0, b3, acc[0][3]);
      acc[1][0] = WMMA_BF16(a1, b0, acc[1][0]);
      acc[1][1] = WMMA_BF16(a1, b1, acc[1][1]);
      acc[1][2] = WMMA_BF16(a1, b2, acc[1][2]);
      acc[1][3] = WMMA_BF16(a1, b3, acc[1][3]);
    }
    __syncthreads();  // protect panel before next stage overwrites it
  }

#pragma unroll
  for (int ti = 0; ti < 2; ++ti) {
#pragma unroll
    for (int tj = 0; tj < 4; ++tj) {
      int n = n0 + tj * 16 + lo;
      int h = n >> 6, d = n & 63;
#pragma unroll
      for (int r = 0; r < 8; ++r) {
        int m  = m0 + ti * 16 + hi * 8 + r;
        int bi = m >> 11, s = m & 2047;
        float v = acc[ti][tj][r] * scale;
        size_t addr;
        if (mode == 0)
          addr = (((size_t)(bi * kNH + h)) * kSeq + s) * kHD + d;
        else
          addr = (((size_t)(bi * kNH + h)) * kHD + d) * kSeq + s;
        outb[addr] = (bf16)v;
      }
    }
  }
}

// ---------------------------------------------------------------------------
// Output GEMM, same async-staged structure: d_out = ao @ Wo^T, fp32 result
// ---------------------------------------------------------------------------
__global__ void __launch_bounds__(256) out_gemm(const bf16* __restrict__ ao,
                                                const bf16* __restrict__ wo,
                                                float* __restrict__ out) {
  __shared__ bf16 apanel[32 * 64];

  int tid  = threadIdx.x;
  int lane = tid & 31;
  int wave = tid >> 5;
  int gw   = blockIdx.x * 8 + wave;
  int m0   = (gw >> 4) * 32;
  int n0   = (gw & 15) * 64;
  int lo = lane & 15, hi = lane >> 4;

  int srow = tid >> 3;
  int scol = (tid & 7) * 8;
  const bf16* sgp = ao + (size_t)(m0 + srow) * kHid + scol;
  unsigned slds = (unsigned)(uintptr_t)(apanel + srow * 64 + scol);

  v8f acc[2][4] = {};
  for (int ks = 0; ks < kHid; ks += 64) {
    async_copy_b128(slds, sgp + ks);
    wait_async0();
    __syncthreads();
#pragma unroll
    for (int kk = 0; kk < 64; kk += 32) {
      v16bf a0 = load_mat_a(apanel, 64, lo,      kk, lane);
      v16bf a1 = load_mat_a(apanel, 64, 16 + lo, kk, lane);
      v16bf b0 = load_mat_b(wo, kHid, n0 + lo,      ks + kk, lane);
      v16bf b1 = load_mat_b(wo, kHid, n0 + 16 + lo, ks + kk, lane);
      v16bf b2 = load_mat_b(wo, kHid, n0 + 32 + lo, ks + kk, lane);
      v16bf b3 = load_mat_b(wo, kHid, n0 + 48 + lo, ks + kk, lane);
      acc[0][0] = WMMA_BF16(a0, b0, acc[0][0]);
      acc[0][1] = WMMA_BF16(a0, b1, acc[0][1]);
      acc[0][2] = WMMA_BF16(a0, b2, acc[0][2]);
      acc[0][3] = WMMA_BF16(a0, b3, acc[0][3]);
      acc[1][0] = WMMA_BF16(a1, b0, acc[1][0]);
      acc[1][1] = WMMA_BF16(a1, b1, acc[1][1]);
      acc[1][2] = WMMA_BF16(a1, b2, acc[1][2]);
      acc[1][3] = WMMA_BF16(a1, b3, acc[1][3]);
    }
    __syncthreads();
  }
#pragma unroll
  for (int ti = 0; ti < 2; ++ti)
#pragma unroll
    for (int tj = 0; tj < 4; ++tj)
#pragma unroll
      for (int r = 0; r < 8; ++r) {
        int m = m0 + ti * 16 + hi * 8 + r;
        out[(size_t)m * kHid + n0 + tj * 16 + lo] = acc[ti][tj][r];
      }
}

// ---------------------------------------------------------------------------
// Sparse attention, flash-style online softmax.
// Each wave owns TWO 16-query tiles so every K/V block load feeds 16 WMMAs.
// q,k: [b,h,s,d] bf16 (q pre-scaled by 1/8); vt: [b,h,d,s] bf16
// ao : [b,s,h*d] bf16
// ---------------------------------------------------------------------------
__global__ void __launch_bounds__(256) sparse_attn(const bf16* __restrict__ qb,
                                                   const bf16* __restrict__ kb,
                                                   const bf16* __restrict__ vt,
                                                   bf16* __restrict__ ao) {
  __shared__ bf16 plds[8][2][16 * 32];  // per-wave, per-qtile P staging (16 KB)

  int lane = threadIdx.x & 31;
  int wave = threadIdx.x >> 5;
  int gw   = blockIdx.x * 8 + wave;   // 0..2047
  int qp   = gw & 63;                 // query pair (32 rows)
  int bh   = gw >> 6;                 // 0..31
  int h    = bh & 15;
  int bi   = bh >> 4;
  bool strided_head = (h < (kNH / 2));
  int qbase = qp * 32;
  int lo = lane & 15, hi = lane >> 4, hi8 = hi * 8;

  const bf16* qh = qb + (size_t)bh * kSeq * kHD;
  const bf16* kh = kb + (size_t)bh * kSeq * kHD;
  const bf16* vh = vt + (size_t)bh * kHD * kSeq;

  // Q A-operands for both tiles (hd = 64 = 2 K-steps), 1/8 scale pre-folded
  v16bf qa[2][2];
#pragma unroll
  for (int t = 0; t < 2; ++t) {
    qa[t][0] = load_mat_a(qh, kHD, qbase + 16 * t + lo, 0,  lane);
    qa[t][1] = load_mat_a(qh, kHD, qbase + 16 * t + lo, 32, lane);
  }

  float mrow[2][8], lrow[2][8];
  v8f o[2][4] = {};
  int imod[2][8], idiv[2][8];  // loop-invariant mask helpers
#pragma unroll
  for (int t = 0; t < 2; ++t)
#pragma unroll
    for (int r = 0; r < 8; ++r) {
      mrow[t][r] = -3.0e38f; lrow[t][r] = 0.0f;
      int i = qbase + 16 * t + hi8 + r;
      imod[t][r] = i % kStride;
      idiv[t][r] = i / kStride;
    }

  int kbmax = (qbase + 31) >> 5;  // causal: skip key blocks fully above diagonal

  for (int kblk = 0; kblk <= kbmax; ++kblk) {
    int kbase = kblk * 32;

    if (kblk < kbmax) {  // prefetch next K block into cache
      __builtin_prefetch(kh + (size_t)(kbase + 32 + lane) * kHD, 0, 3);
    }

    // shared K B-operands for this key block
    v16bf b00 = load_mat_b(kh, kHD, kbase + lo,      0,  lane);
    v16bf b01 = load_mat_b(kh, kHD, kbase + lo,      32, lane);
    v16bf b10 = load_mat_b(kh, kHD, kbase + 16 + lo, 0,  lane);
    v16bf b11 = load_mat_b(kh, kHD, kbase + 16 + lo, 32, lane);

    // per-block (lane-only) mask helpers
    int j0 = kbase + lo, j1 = j0 + 16;
    int j0m = j0 % kStride, j0d = j0 / kStride;
    int j1m = j1 % kStride, j1d = j1 / kStride;

#pragma unroll
    for (int t = 0; t < 2; ++t) {
      v8f c0 = {}, c1 = {};
      c0 = WMMA_BF16(qa[t][0], b00, c0);
      c0 = WMMA_BF16(qa[t][1], b01, c0);
      c1 = WMMA_BF16(qa[t][0], b10, c1);
      c1 = WMMA_BF16(qa[t][1], b11, c1);

      bf16* myp = plds[wave][t];
#pragma unroll
      for (int r = 0; r < 8; ++r) {
        int i = qbase + 16 * t + hi8 + r;
        bool ok0, ok1;
        if (strided_head) {
          ok0 = (j0 <= i) && ((i - j0 < kStride) || (j0m == imod[t][r]));
          ok1 = (j1 <= i) && ((i - j1 < kStride) || (j1m == imod[t][r]));
        } else {
          ok0 = (j0 <= i) && ((j0d == idiv[t][r]) || (j0m >= kStride - 1));
          ok1 = (j1 <= i) && ((j1d == idiv[t][r]) || (j1m >= kStride - 1));
        }
        float s0 = ok0 ? c0[r] : -3.0e38f;
        float s1 = ok1 ? c1[r] : -3.0e38f;
        float tm = fmaxf(s0, s1);
        tm = fmaxf(tm, __shfl_xor(tm, 1, 32));
        tm = fmaxf(tm, __shfl_xor(tm, 2, 32));
        tm = fmaxf(tm, __shfl_xor(tm, 4, 32));
        tm = fmaxf(tm, __shfl_xor(tm, 8, 32));
        float nm    = fmaxf(mrow[t][r], tm);
        float alpha = __expf(mrow[t][r] - nm);
        float e0 = ok0 ? __expf(s0 - nm) : 0.0f;
        float e1 = ok1 ? __expf(s1 - nm) : 0.0f;
        float rs = e0 + e1;
        rs += __shfl_xor(rs, 1, 32);
        rs += __shfl_xor(rs, 2, 32);
        rs += __shfl_xor(rs, 4, 32);
        rs += __shfl_xor(rs, 8, 32);
        lrow[t][r] = lrow[t][r] * alpha + rs;
        mrow[t][r] = nm;
        o[t][0][r] *= alpha; o[t][1][r] *= alpha;
        o[t][2][r] *= alpha; o[t][3][r] *= alpha;
        myp[(hi8 + r) * 32 + lo]      = (bf16)e0;
        myp[(hi8 + r) * 32 + 16 + lo] = (bf16)e1;
      }
    }
    asm volatile("" ::: "memory");  // keep P loads after P stores (DS in-order per wave)

    // reload P tiles in A-operand layout
    v16bf pa[2];
#pragma unroll
    for (int t = 0; t < 2; ++t) {
      const bf16* prow = plds[wave][t] + lo * 32;
      int sel = hi8;
      v8bf pc0 = *(const v8bf*)(prow + sel);
      v8bf pc1 = *(const v8bf*)(prow + 16 + sel);
#pragma unroll
      for (int u = 0; u < 8; ++u) { pa[t][u] = pc0[u]; pa[t][8 + u] = pc1[u]; }
    }
    asm volatile("" ::: "memory");

    // shared V B-operands (transposed V: contiguous), feed both query tiles
    v16bf vb0 = load_mat_b(vh, kSeq, 0 * 16 + lo, kbase, lane);
    v16bf vb1 = load_mat_b(vh, kSeq, 1 * 16 + lo, kbase, lane);
    v16bf vb2 = load_mat_b(vh, kSeq, 2 * 16 + lo, kbase, lane);
    v16bf vb3 = load_mat_b(vh, kSeq, 3 * 16 + lo, kbase, lane);
#pragma unroll
    for (int t = 0; t < 2; ++t) {
      o[t][0] = WMMA_BF16(pa[t], vb0, o[t][0]);
      o[t][1] = WMMA_BF16(pa[t], vb1, o[t][1]);
      o[t][2] = WMMA_BF16(pa[t], vb2, o[t][2]);
      o[t][3] = WMMA_BF16(pa[t], vb3, o[t][3]);
    }
  }

  // normalize + store merged-head bf16 [b, s, h*64 + d]
#pragma unroll
  for (int t = 0; t < 2; ++t)
#pragma unroll
    for (int r = 0; r < 8; ++r) {
      float inv = 1.0f / lrow[t][r];
      int s = qbase + 16 * t + hi8 + r;
      size_t row = ((size_t)(bi * kSeq + s)) * kHid + (size_t)h * kHD;
      ao[row +  0 + lo] = (bf16)(o[t][0][r] * inv);
      ao[row + 16 + lo] = (bf16)(o[t][1][r] * inv);
      ao[row + 32 + lo] = (bf16)(o[t][2][r] * inv);
      ao[row + 48 + lo] = (bf16)(o[t][3][r] * inv);
    }
}

// ---------------------------------------------------------------------------
// Host-side orchestration
// ---------------------------------------------------------------------------
extern "C" void kernel_launch(void* const* d_in, const int* in_sizes, int n_in,
                              void* d_out, int out_size, void* d_ws, size_t ws_size,
                              hipStream_t stream) {
  (void)in_sizes; (void)n_in; (void)out_size; (void)ws_size;
  const float* x  = (const float*)d_in[0];
  const float* Wq = (const float*)d_in[1];
  const float* Wk = (const float*)d_in[2];
  const float* Wv = (const float*)d_in[3];
  const float* Wo = (const float*)d_in[4];
  float* out = (float*)d_out;

  bf16* ws = (bf16*)d_ws;
  bf16* xb  = ws;                       // [4096,1024]
  bf16* wqb = xb  + kXN;
  bf16* wkb = wqb + kWN;
  bf16* wvb = wkb + kWN;
  bf16* wob = wvb + kWN;
  bf16* qbf = wob + kWN;                // [b,h,s,d]
  bf16* kbf = qbf + kXN;                // [b,h,s,d]
  bf16* vtb = kbf + kXN;                // [b,h,d,s]
  bf16* aob = vtb + kXN;                // [b,s,h*d]

  // stage bf16 copies
  cvt_f32_bf16<<<2048, 256, 0, stream>>>(x,  xb,  (int)kXN);
  cvt_f32_bf16<<<1024, 256, 0, stream>>>(Wq, wqb, (int)kWN);
  cvt_f32_bf16<<<1024, 256, 0, stream>>>(Wk, wkb, (int)kWN);
  cvt_f32_bf16<<<1024, 256, 0, stream>>>(Wv, wvb, (int)kWN);
  cvt_f32_bf16<<<1024, 256, 0, stream>>>(Wo, wob, (int)kWN);

  // projections: 128 x 16 supertiles = 2048 wave-jobs / 8 waves = 256 WGs
  proj_gemm<<<256, 256, 0, stream>>>(xb, wqb, qbf, 0, 0.125f);  // fold hd^-0.5 into Q
  proj_gemm<<<256, 256, 0, stream>>>(xb, wkb, kbf, 0, 1.0f);
  proj_gemm<<<256, 256, 0, stream>>>(xb, wvb, vtb, 1, 1.0f);    // V transposed

  // attention: 32 (b,h) x 64 query-pairs = 2048 wave-jobs / 8 = 256 WGs
  sparse_attn<<<256, 256, 0, stream>>>(qbf, kbf, vtb, aob);

  // final projection, fp32 output
  out_gemm<<<256, 256, 0, stream>>>(aob, wob, out);
}
